// RWKV_Tmix_x052_45732811768066
// MI455X (gfx1250) — matrix operand order repaired and checked by reading, not compiled, not run on hardware
//
#include <hip/hip_runtime.h>

#define Bsz 4
#define Tn  4096
#define Cn  1024
#define Hn  64
#define Nn  16
#define MT  (Bsz * Tn)   // 16384 rows

typedef __attribute__((ext_vector_type(16))) __bf16 v16bf;
typedef __attribute__((ext_vector_type(8)))  float  v8f;

union BfFrag { uint4 q[2]; v16bf v; };

__device__ __forceinline__ unsigned short f2bf(float f) {
    unsigned int u = __float_as_uint(f);
    unsigned int r = u + 0x7FFFu + ((u >> 16) & 1u);   // RNE
    return (unsigned short)(r >> 16);
}
__device__ __forceinline__ float bf2f(unsigned short h) {
    return __uint_as_float(((unsigned int)h) << 16);
}

// ---------------------------------------------------------------------------
// Kernel 1: fp32 (K,N) weight -> bf16 (N,K) transposed weight
// ---------------------------------------------------------------------------
__global__ __launch_bounds__(256) void wconv_kernel(const float* __restrict__ W,
                                                    unsigned short* __restrict__ Wt) {
    int idx = blockIdx.x * 256 + threadIdx.x;     // over Cn*Cn
    int n = idx / Cn;
    int k = idx - n * Cn;
    Wt[(size_t)n * Cn + k] = f2bf(W[(size_t)k * Cn + n]);
}

// ---------------------------------------------------------------------------
// Kernel 2: token shift + LoRA mixing -> xk/xv/xr/xg (bf16). One block/token.
// ---------------------------------------------------------------------------
__global__ __launch_bounds__(256) void mix_kernel(
    const float* __restrict__ x,
    const float* __restrict__ tmx,
    const float* __restrict__ tmk,
    const float* __restrict__ tmv,
    const float* __restrict__ tmr,
    const float* __restrict__ tmg,
    const float* __restrict__ w1,    // (C, 80)
    const float* __restrict__ w2,    // (5, 16, C) flat
    unsigned short* __restrict__ Xk,
    unsigned short* __restrict__ Xv,
    unsigned short* __restrict__ Xr,
    unsigned short* __restrict__ Xg)
{
    __shared__ float xs[Cn];
    __shared__ float xxs[Cn];
    __shared__ float xxxs[Cn];
    __shared__ float pm[2][80];
    __shared__ float mm[80];

    const int bt  = blockIdx.x;          // b*T + t
    const int t   = bt & (Tn - 1);
    const int tid = threadIdx.x;

    const float* xrow  = x + (size_t)bt * Cn;
    float4 xv4 = ((const float4*)xrow)[tid];
    float4 xp4 = make_float4(0.f, 0.f, 0.f, 0.f);
    if (t > 0) xp4 = ((const float4*)(xrow - Cn))[tid];

    int c0 = tid * 4;
    float xv_[4] = {xv4.x, xv4.y, xv4.z, xv4.w};
    float xp_[4] = {xp4.x, xp4.y, xp4.z, xp4.w};
#pragma unroll
    for (int i = 0; i < 4; ++i) {
        int c = c0 + i;
        float xc  = xv_[i];
        float xxc = xp_[i] - xc;         // shifted - x
        xs[c]   = xc;
        xxs[c]  = xxc;
        xxxs[c] = xc + xxc * tmx[c];
    }
    __syncthreads();

    // xxx @ w1 -> tanh, split each 1024-dot over two threads
    if (tid < 160) {
        int o    = tid % 80;
        int half = tid / 80;
        int cb   = half * 512;
        float s = 0.f;
        for (int c = cb; c < cb + 512; ++c) s += xxxs[c] * w1[(size_t)c * 80 + o];
        pm[half][o] = s;
    }
    __syncthreads();
    if (tid < 80) mm[tid] = tanhf(pm[0][tid] + pm[1][tid]);
    __syncthreads();

    // project m (f=1..4; f=0 / mw is unused in the reference) and form xk/xv/xr/xg
    for (int c = tid; c < Cn; c += 256) {
        float mk = 0.f, mv = 0.f, mr = 0.f, mg = 0.f;
#pragma unroll
        for (int kk = 0; kk < 16; ++kk) {
            mk += mm[16 + kk] * w2[(size_t)(16 + kk) * Cn + c];
            mv += mm[32 + kk] * w2[(size_t)(32 + kk) * Cn + c];
            mr += mm[48 + kk] * w2[(size_t)(48 + kk) * Cn + c];
            mg += mm[64 + kk] * w2[(size_t)(64 + kk) * Cn + c];
        }
        float xc = xs[c], xxc = xxs[c];
        size_t idx = (size_t)bt * Cn + c;
        Xk[idx] = f2bf(xc + xxc * (tmk[c] + mk));
        Xv[idx] = f2bf(xc + xxc * (tmv[c] + mv));
        Xr[idx] = f2bf(xc + xxc * (tmr[c] + mr));
        Xg[idx] = f2bf(xc + xxc * (tmg[c] + mg));
    }
}

// ---------------------------------------------------------------------------
// Kernel 3: bf16 WMMA GEMM  Y(M,1024) = A(M,1024) @ W(1024,1024)
// A row-major bf16, Bt = W^T (N,K) row-major bf16.
// Block = 8 waves, all sharing one 64-wide N strip. B slab staged in LDS
// (double-buffered, padded rows). Each wave computes a 32(M) x 64(N) tile:
// 2 A fragments x 4 B fragments = 8 WMMAs per 32-deep k-step.
// modes: 0 = bf16 out, 1 = silu -> bf16, 2 = f32 out
// ---------------------------------------------------------------------------
#define BS_STRIDE 56   // halves per LDS row (112B: 16B-aligned, 28-bank stride)

__global__ __launch_bounds__(256) void gemm_bf16_kernel(
    const unsigned short* __restrict__ A,
    const unsigned short* __restrict__ Bt,
    void* __restrict__ Out, int mode)
{
    const int K = Cn, Nw = Cn;
    const int lane  = threadIdx.x;          // 0..31
    const int wave  = threadIdx.y;          // 0..7
    const int mTile = blockIdx.y * 8 + wave;   // 32-row tile index
    const int nTile = blockIdx.x;              // 64-col tile index
    const int l15   = lane & 15;
    const int hi    = lane >> 4;

    __shared__ __align__(16) unsigned short Bs[2][64 * BS_STRIDE];

    // --- B staging: 256 threads move the 64x32 bf16 slab (4KB) per k-step ---
    const int tid    = wave * 32 + lane;
    const int sNrel  = tid >> 2;            // 0..63  (column within strip)
    const int sChunk = (tid & 3) * 8;       // 0,8,16,24 (halves within k-slab)
    const unsigned short* sSrc = Bt + (size_t)(nTile * 64 + sNrel) * K + sChunk;
    const int sDst = sNrel * BS_STRIDE + sChunk;

    // --- A fragment bases: lanes<16 read K {0..7,16..23}, lanes>=16 offset +8
    const unsigned short* aRow0 = A + (size_t)(mTile * 32 + l15) * K + (hi ? 8 : 0);
    const unsigned short* aRow1 = aRow0 + (size_t)16 * K;

    // --- B fragment LDS offsets: lane holds col n=s*16+l15, K {0..7,16..23} + 8*hi ---
    const int bOff[4] = { (0 + l15) * BS_STRIDE + hi * 8,
                          (16 + l15) * BS_STRIDE + hi * 8,
                          (32 + l15) * BS_STRIDE + hi * 8,
                          (48 + l15) * BS_STRIDE + hi * 8 };

    v8f acc0[4], acc1[4];
#pragma unroll
    for (int s = 0; s < 4; ++s) {
        acc0[s] = (v8f){0.f,0.f,0.f,0.f,0.f,0.f,0.f,0.f};
        acc1[s] = (v8f){0.f,0.f,0.f,0.f,0.f,0.f,0.f,0.f};
    }

    // prologue: stage first slab
    *(uint4*)&Bs[0][sDst] = *(const uint4*)(sSrc);
    __syncthreads();

    for (int k0 = 0; k0 < K; k0 += 32) {
        const int cur = (k0 >> 5) & 1;
        if (k0 + 32 < K)   // stage next slab into the other buffer
            *(uint4*)&Bs[cur ^ 1][sDst] = *(const uint4*)(sSrc + k0 + 32);

        BfFrag a0, a1;
        a0.q[0] = *(const uint4*)(aRow0 + k0);
        a0.q[1] = *(const uint4*)(aRow0 + k0 + 16);
        a1.q[0] = *(const uint4*)(aRow1 + k0);
        a1.q[1] = *(const uint4*)(aRow1 + k0 + 16);

        const unsigned short* bs = Bs[cur];
#pragma unroll
        for (int s = 0; s < 4; ++s) {
            BfFrag b;
            b.q[0] = *(const uint4*)(bs + bOff[s]);
            b.q[1] = *(const uint4*)(bs + bOff[s] + 16);
            acc0[s] = __builtin_amdgcn_wmma_f32_16x16x32_bf16(false, a0.v, false, b.v,
                                                             (short)0, acc0[s], false, false);
            acc1[s] = __builtin_amdgcn_wmma_f32_16x16x32_bf16(false, a1.v, false, b.v,
                                                             (short)0, acc1[s], false, false);
        }
        __syncthreads();   // all reads of cur done; nxt fully staged
    }

    // C/D layout: lane l -> N = l%16, M = r + 8*(l>=16)
#pragma unroll
    for (int s = 0; s < 4; ++s) {
        int n = nTile * 64 + s * 16 + l15;
#pragma unroll
        for (int r = 0; r < 8; ++r) {
            int m0 = mTile * 32 + r + hi * 8;
            int m1 = m0 + 16;
            float v0 = acc0[s][r];
            float v1 = acc1[s][r];
            if (mode == 2) {
                ((float*)Out)[(size_t)m0 * Nw + n] = v0;
                ((float*)Out)[(size_t)m1 * Nw + n] = v1;
            } else {
                if (mode == 1) {
                    v0 = v0 / (1.f + expf(-v0));   // silu
                    v1 = v1 / (1.f + expf(-v1));
                }
                ((unsigned short*)Out)[(size_t)m0 * Nw + n] = f2bf(v0);
                ((unsigned short*)Out)[(size_t)m1 * Nw + n] = f2bf(v1);
            }
        }
    }
}

// ---------------------------------------------------------------------------
// Kernel 4: WKV5 recurrence + GroupNorm + g-gating.
// One wave handles 2 (b,h) head slots (16 lanes each). Lane j keeps S[0..15][j].
// ---------------------------------------------------------------------------
__global__ __launch_bounds__(32) void wkv_gn_kernel(
    const unsigned short* __restrict__ R,
    const unsigned short* __restrict__ Kb,
    const unsigned short* __restrict__ V,
    const unsigned short* __restrict__ G,
    const float* __restrict__ tdecay,   // (H,N)
    const float* __restrict__ tfaaaa,   // (H,N)
    const float* __restrict__ gnw,
    const float* __restrict__ gnb,
    unsigned short* __restrict__ OG)
{
    const int lane = threadIdx.x;
    const int slot = blockIdx.x * 2 + (lane >> 4);   // 0..255 = b*H + h
    const int b    = slot >> 6;
    const int h    = slot & 63;
    const int j    = lane & 15;
    const int c    = h * 16 + j;

    const float w = expf(-expf(bf2f(f2bf(tdecay[c]))));  // bf16 round-trip like ref
    const float u = bf2f(f2bf(tfaaaa[c]));
    const float gw = gnw[c], gb = gnb[c];

    float wArr[16];
#pragma unroll
    for (int i = 0; i < 16; ++i) wArr[i] = __shfl(w, (lane & 16) | i, 32);

    float S[16];
#pragma unroll
    for (int i = 0; i < 16; ++i) S[i] = 0.f;

    size_t base = (size_t)b * Tn * Cn + h * 16 + j;
    for (int t = 0; t < Tn; ++t, base += Cn) {
        float rl = bf2f(R[base]);
        float kl = bf2f(Kb[base]);
        float vl = bf2f(V[base]);

        // sum_i r_i*u_i*k_i   (xor masks <= 8 stay inside each 16-lane half)
        float p = rl * u * kl;
        p += __shfl_xor(p, 1, 32); p += __shfl_xor(p, 2, 32);
        p += __shfl_xor(p, 4, 32); p += __shfl_xor(p, 8, 32);
        float y = p * vl;

#pragma unroll
        for (int i = 0; i < 16; ++i) {
            int src = (lane & 16) | i;
            float ri = __shfl(rl, src, 32);
            float ki = __shfl(kl, src, 32);
            y += ri * S[i];
            S[i] = wArr[i] * S[i] + ki * vl;
        }

        // GroupNorm over the head's 16 channels (these 16 lanes)
        float yb = bf2f(f2bf(y));           // ref casts y to bf16 first
        float xn = yb * 0.125f;             // / HEAD_SIZE_DIVISOR
        float s1 = xn, s2 = xn * xn;
        s1 += __shfl_xor(s1, 1, 32); s1 += __shfl_xor(s1, 2, 32);
        s1 += __shfl_xor(s1, 4, 32); s1 += __shfl_xor(s1, 8, 32);
        s2 += __shfl_xor(s2, 1, 32); s2 += __shfl_xor(s2, 2, 32);
        s2 += __shfl_xor(s2, 4, 32); s2 += __shfl_xor(s2, 8, 32);
        float mean = s1 * (1.f / 16.f);
        float var  = s2 * (1.f / 16.f) - mean * mean;
        float o    = (xn - mean) * rsqrtf(var + 1e-5f) * gw + gb;

        float gf = bf2f(G[base]);
        OG[base] = f2bf(o * gf);
    }
}

// ---------------------------------------------------------------------------
extern "C" void kernel_launch(void* const* d_in, const int* in_sizes, int n_in,
                              void* d_out, int out_size, void* d_ws, size_t ws_size,
                              hipStream_t stream) {
    const float* x      = (const float*)d_in[0];
    const float* tmx    = (const float*)d_in[1];
    // d_in[2] time_mix_w: unused (xw is dead in the reference)
    const float* tmk    = (const float*)d_in[3];
    const float* tmv    = (const float*)d_in[4];
    const float* tmr    = (const float*)d_in[5];
    const float* tmg    = (const float*)d_in[6];
    const float* w1     = (const float*)d_in[7];
    const float* w2     = (const float*)d_in[8];
    const float* tdecay = (const float*)d_in[9];
    const float* tfaaaa = (const float*)d_in[10];
    const float* Wr     = (const float*)d_in[11];
    const float* Wk     = (const float*)d_in[12];
    const float* Wv     = (const float*)d_in[13];
    const float* Wg     = (const float*)d_in[14];
    const float* Wo     = (const float*)d_in[15];
    const float* gnw    = (const float*)d_in[16];
    const float* gnb    = (const float*)d_in[17];
    float* out = (float*)d_out;

    // workspace layout
    char* ws = (char*)d_ws;
    const size_t xbytes = (size_t)MT * Cn * sizeof(unsigned short);   // 32 MB
    const size_t wbytes = (size_t)Cn * Cn * sizeof(unsigned short);   //  2 MB
    unsigned short* Xk = (unsigned short*)ws; ws += xbytes;
    unsigned short* Xv = (unsigned short*)ws; ws += xbytes;
    unsigned short* Xr = (unsigned short*)ws; ws += xbytes;
    unsigned short* Xg = (unsigned short*)ws; ws += xbytes;
    unsigned short* Rr = (unsigned short*)ws; ws += xbytes;
    unsigned short* Kk = (unsigned short*)ws; ws += xbytes;
    unsigned short* Vv = (unsigned short*)ws; ws += xbytes;
    unsigned short* Gg = (unsigned short*)ws; ws += xbytes;
    unsigned short* WtR = (unsigned short*)ws; ws += wbytes;
    unsigned short* WtK = (unsigned short*)ws; ws += wbytes;
    unsigned short* WtV = (unsigned short*)ws; ws += wbytes;
    unsigned short* WtG = (unsigned short*)ws; ws += wbytes;
    unsigned short* WtO = (unsigned short*)ws; ws += wbytes;
    unsigned short* OG  = Xk;   // Xk is dead after the k-projection GEMM

    // 1) weights -> bf16 transposed
    const int wgrid = (Cn * Cn) / 256;
    wconv_kernel<<<wgrid, 256, 0, stream>>>(Wr, WtR);
    wconv_kernel<<<wgrid, 256, 0, stream>>>(Wk, WtK);
    wconv_kernel<<<wgrid, 256, 0, stream>>>(Wv, WtV);
    wconv_kernel<<<wgrid, 256, 0, stream>>>(Wg, WtG);
    wconv_kernel<<<wgrid, 256, 0, stream>>>(Wo, WtO);

    // 2) token shift + LoRA mix
    mix_kernel<<<MT, 256, 0, stream>>>(x, tmx, tmk, tmv, tmr, tmg, w1, w2,
                                       Xk, Xv, Xr, Xg);

    // 3) r/k/v/g projections (bf16 WMMA, LDS-staged B)
    dim3 gblock(32, 8);
    dim3 ggrid(Cn / 64, MT / 256);           // 32 M-rows per wave, 8 waves/block
    gemm_bf16_kernel<<<ggrid, gblock, 0, stream>>>(Xr, WtR, Rr, 0);
    gemm_bf16_kernel<<<ggrid, gblock, 0, stream>>>(Xk, WtK, Kk, 0);
    gemm_bf16_kernel<<<ggrid, gblock, 0, stream>>>(Xv, WtV, Vv, 0);
    gemm_bf16_kernel<<<ggrid, gblock, 0, stream>>>(Xg, WtG, Gg, 1);  // silu

    // 4) WKV5 recurrence + GroupNorm + gate
    wkv_gn_kernel<<<(Bsz * Hn) / 2, 32, 0, stream>>>(Rr, Kk, Vv, Gg,
                                                     tdecay, tfaaaa, gnw, gnb, OG);

    // 5) output projection (f32 out)
    gemm_bf16_kernel<<<ggrid, gblock, 0, stream>>>(OG, WtO, out, 2);
}
